// MyCustomLayer_48180943127175
// MI455X (gfx1250) — matrix operand order repaired and checked by reading, not compiled, not run on hardware
//
#include <hip/hip_runtime.h>
#include <hip/hip_bf16.h>

typedef _Float16 v16h __attribute__((ext_vector_type(16)));
typedef _Float16 h8   __attribute__((ext_vector_type(8)));
typedef float    v8f  __attribute__((ext_vector_type(8)));

#define BATCH 128
#define NLM   98     // landmarks
#define NPTS  196
#define WID   128
#define HEI   128
#define KPAD  128    // landmarks padded to multiple of 32 for WMMA K-steps

// out[b][h][w] = sum_k exp(-(h-y_k)^2 * c) * exp(-(w-x_k)^2 * c)
//             = (Gy^T x Gx)[h][w]  -- per-batch 128x128x(98->128) GEMM on WMMA.
__global__ __launch_bounds__(256) void heatmap_sep_wmma(const float* __restrict__ X,
                                                        float* __restrict__ out)
{
    __shared__ _Float16 sGy[HEI * KPAD]; // [h][k]  (K-major per column)
    __shared__ _Float16 sGx[WID * KPAD]; // [w][k]
    __shared__ float    sLM[NPTS];

    const int b = blockIdx.x;
    const int t = threadIdx.x;

    if (t < NPTS) sLM[t] = X[b * NPTS + t];
    __syncthreads();

    // ---- Phase 1: separable 1-D Gaussian tables in f16 -------------------
    // 2048 work items = col(0..127) x kgroup(0..15); 8 k's per item, one
    // ds_store_b128 per table per item (vs 8x b16 stores).
    {
        const float C2 = 0.028853900817779268f; // log2(e) / (2*sigma^2), sigma=5
        #pragma unroll
        for (int i = 0; i < 8; ++i) {
            const int g   = i * 256 + t;
            const int col = g >> 4;
            const int kg  = (g & 15) * 8;
            const float fc = (float)col;
            h8 vx, vy;
            #pragma unroll
            for (int j = 0; j < 8; ++j) {
                const int k   = kg + j;
                const bool ok = (k < NLM);
                const int kc  = ok ? k : 0;          // clamp: no OOB LDS read
                float dx = fc - sLM[2 * kc];
                float dy = fc - sLM[2 * kc + 1];
                float gx = __builtin_amdgcn_exp2f(-(dx * dx) * C2); // v_exp_f32
                float gy = __builtin_amdgcn_exp2f(-(dy * dy) * C2);
                vx[j] = (_Float16)(ok ? gx : 0.f);   // v_cndmask, branchless pad
                vy[j] = (_Float16)(ok ? gy : 0.f);
            }
            *(h8*)(&sGx[col * KPAD + kg]) = vx;
            *(h8*)(&sGy[col * KPAD + kg]) = vy;
        }
    }
    __syncthreads();

    // ---- Phase 2: WMMA GEMM  D(16x16) += A(16x32) x B(32x16) ------------
    const int wave = t >> 5;
    const int lane = t & 31;
    const int h0   = wave * 16;          // each wave owns one 16-row band
    const int m    = lane & 15;          // A: row index
    const int kbA  = (lane >> 4) * 8;    // A 16-bit layout: hi half-wave -> K+8
    const int n    = lane & 15;          // B: column index
    const int kbB  = (lane >> 4) * 8;    // B 16-bit layout: hi half-wave -> K+8

    // A fragments (Gy^T tile rows h0..h0+15), 4 K-steps, hoisted across tiles.
    v16h A[4];
    const _Float16* gyRow = &sGy[(h0 + m) * KPAD];
    #pragma unroll
    for (int kk = 0; kk < 4; ++kk) {
        union { v16h v; h8 h[2]; } u;
        u.h[0] = *(const h8*)(gyRow + kk * 32 + kbA);        // K = kk*32+kbA .. +7
        u.h[1] = *(const h8*)(gyRow + kk * 32 + kbA + 16);   // K = +16 .. +23
        A[kk] = u.v;
    }

    float* outB = out + (size_t)b * (HEI * WID);
    const int mBase = (lane >> 4) * 8;   // C/D layout: lanes 16-31 hold M=8..15

    #pragma unroll
    for (int w0 = 0; w0 < WID; w0 += 16) {
        v8f c = {};
        const _Float16* gxRow = &sGx[(w0 + n) * KPAD];
        #pragma unroll
        for (int kk = 0; kk < 4; ++kk) {
            union { v16h v; h8 h[2]; } u;
            u.h[0] = *(const h8*)(gxRow + kk * 32 + kbB);     // K = kk*32+kbB .. +7
            u.h[1] = *(const h8*)(gxRow + kk * 32 + kbB + 16); // K = +16 .. +23
            c = __builtin_amdgcn_wmma_f32_16x16x32_f16(
                    /*neg_a=*/false, A[kk], /*neg_b=*/false, u.v,
                    /*c_mod=*/(short)0, c, /*reuse_a=*/false, /*reuse_b=*/false);
        }
        #pragma unroll
        for (int r = 0; r < 8; ++r) {
            outB[(h0 + mBase + r) * WID + (w0 + n)] = c[r];
        }
    }
}

extern "C" void kernel_launch(void* const* d_in, const int* in_sizes, int n_in,
                              void* d_out, int out_size, void* d_ws, size_t ws_size,
                              hipStream_t stream) {
    const float* X = (const float*)d_in[0];
    float* out = (float*)d_out;
    heatmap_sep_wmma<<<BATCH, 256, 0, stream>>>(X, out);
}
